// LANDER_57853209477715
// MI455X (gfx1250) — matrix operand order repaired and checked by reading, not compiled, not run on hardware
//
#include <hip/hip_runtime.h>
#include <hip/hip_bf16.h>
#include <stdint.h>

// ---------------------------------------------------------------------------
// Types for CDNA5 WMMA (wave32): v_wmma_f32_16x16x32_bf16
// ---------------------------------------------------------------------------
typedef __attribute__((ext_vector_type(16))) __bf16 v16bf;
typedef __attribute__((ext_vector_type(8)))  float  v8f;
typedef __attribute__((ext_vector_type(4)))  int    v4i;

#define NN 50000
#define EE 800000

// CDNA5 async global->LDS path (ASYNCcnt-tracked), if the toolchain has it.
// Probe-confirmed signature: (v4i addrspace(1)*, v4i addrspace(3)*, Imm, Imm)
#if __has_builtin(__builtin_amdgcn_global_load_async_to_lds_b128)
#define HAVE_ASYNC_LDS 1
typedef __attribute__((address_space(1))) v4i g_v4i;
typedef __attribute__((address_space(3))) v4i l_v4i;
#endif

// ---------------------------------------------------------------------------
// Helpers
// ---------------------------------------------------------------------------
__device__ inline unsigned short f2bf(float f) {
  unsigned int u = __float_as_uint(f);
  u += 0x7fffu + ((u >> 16) & 1u);      // round-to-nearest-even
  return (unsigned short)(u >> 16);
}
__device__ inline int f2ord(float f) {   // monotone float->int mapping
  int i = __float_as_int(f);
  return (i >= 0) ? i : (i ^ 0x7fffffff);
}
__device__ inline float ord2f(int i) {
  return __int_as_float((i >= 0) ? i : (i ^ 0x7fffffff));
}
__device__ inline float atomAddF(float* p, float v) {
  // relaxed, agent scope -> native global_atomic_add_f32 on gfx1250
  return __hip_atomic_fetch_add(p, v, __ATOMIC_RELAXED, __HIP_MEMORY_SCOPE_AGENT);
}
__device__ inline void async_wait_all() {
#ifdef HAVE_ASYNC_LDS
#if __has_builtin(__builtin_amdgcn_s_wait_asynccnt)
  __builtin_amdgcn_s_wait_asynccnt(0);
#else
  asm volatile("s_wait_asynccnt 0x0" ::: "memory");
#endif
#endif
}

// ---------------------------------------------------------------------------
// WMMA GEMM: C[M,O] = A[M,K](bf16) * B[K,O](bf16) (+bias, optional relu)
// Block: 256 threads = 8 waves. Block tile 128x64, wave tile 32x32 (2x2 WMMA),
// K-step 32. Double-buffered LDS (one barrier per K-step); A tile staged with
// GLOBAL_LOAD_ASYNC_TO_LDS_B128 when available; B staged transposed [n][k] so
// fragments are plain b128 LDS loads.
// ---------------------------------------------------------------------------
#define BM 128
#define BN 64
#define BK 32
#define LDSA 40   // padded row stride (elems); *2B = 80B keeps 16B alignment
#define LDSB 40

union FragU { uint4 u[2]; v16bf v; };

__global__ __launch_bounds__(256) void k_gemm_bf16(
    const unsigned short* __restrict__ A,
    const unsigned short* __restrict__ B,
    const float* __restrict__ bias,
    float* __restrict__ C,
    int M, int K, int O, int relu)
{
  __shared__ unsigned short lsA[2][BM * LDSA];
  __shared__ unsigned short lsB[2][BN * LDSB];

  const int tid  = threadIdx.x;
  const int lane = tid & 31;
  const int wave = tid >> 5;
  const int wm = (wave & 3) * 32;       // wave row offset inside block tile
  const int wn = (wave >> 2) * 32;      // wave col offset
  const int bm = blockIdx.x * BM;
  const int bn = blockIdx.y * BN;

  v8f acc[2][2];
#pragma unroll
  for (int i = 0; i < 2; ++i)
#pragma unroll
    for (int j = 0; j < 2; ++j)
#pragma unroll
      for (int e = 0; e < 8; ++e) acc[i][j][e] = 0.0f;

  // global->LDS A mapping: 2 threads/row, 16 cols each (two 16B chunks)
  const int aRow = tid >> 1;
  const int aCol = (tid & 1) * 16;
  // global->LDS B mapping: thread covers fixed n, 8 consecutive k (transpose)
  const int bN = tid & 63;
  const int bK = (tid >> 6) * 8;

  int gr = bm + aRow; if (gr >= M) gr = M - 1;   // clamp; junk rows never stored

  const int laneRow = lane & 15;
  const int aKlo = (lane < 16) ? 0 : 16;   // A frag: K 0-15 vs 16-31
  const int bKh  = (lane < 16) ? 0 : 16;   // B frag: K 0-15 vs 16-31

  // Stage one BKxBM A tile + BKxBN B tile (transposed) into LDS buffer `buf`.
  auto stage = [&](int k0, int buf) {
    const unsigned short* gA = A + (size_t)gr * K + (k0 + aCol);
    unsigned short* la = &lsA[buf][aRow * LDSA + aCol];
#ifdef HAVE_ASYNC_LDS
    __builtin_amdgcn_global_load_async_to_lds_b128(
        (g_v4i*)gA, (l_v4i*)la, 0, 0);
    __builtin_amdgcn_global_load_async_to_lds_b128(
        (g_v4i*)(gA + 8), (l_v4i*)(la + 8), 0, 0);
#else
    uint4 av0 = ((const uint4*)gA)[0];
    uint4 av1 = ((const uint4*)gA)[1];
    *(uint4*)la       = av0;
    *(uint4*)(la + 8) = av1;
#endif
    __align__(16) unsigned short tb[8];
#pragma unroll
    for (int j = 0; j < 8; ++j)
      tb[j] = B[(size_t)(k0 + bK + j) * O + (bn + bN)];   // coalesced per j
    *(uint4*)&lsB[buf][bN * LDSB + bK] = *(const uint4*)tb;
  };

  const int nk = K / BK;
  stage(0, 0);
  async_wait_all();
  __syncthreads();

  for (int t = 0; t < nk; ++t) {
    const int buf = t & 1;
    if (t + 1 < nk) stage((t + 1) * BK, buf ^ 1);   // prefetch next tile

    FragU fa[2], fb[2];
#pragma unroll
    for (int i = 0; i < 2; ++i) {
      int r = wm + i * 16 + laneRow;
      fa[i].u[0] = *(const uint4*)&lsA[buf][r * LDSA + aKlo];
      fa[i].u[1] = *(const uint4*)&lsA[buf][r * LDSA + aKlo + 8];
    }
#pragma unroll
    for (int j = 0; j < 2; ++j) {
      int c = wn + j * 16 + laneRow;
      fb[j].u[0] = *(const uint4*)&lsB[buf][c * LDSB + bKh];
      fb[j].u[1] = *(const uint4*)&lsB[buf][c * LDSB + bKh + 8];
    }

#pragma unroll
    for (int i = 0; i < 2; ++i)
#pragma unroll
      for (int j = 0; j < 2; ++j)
        acc[i][j] = __builtin_amdgcn_wmma_f32_16x16x32_bf16(
            false, fa[i].v, false, fb[j].v, (short)0, acc[i][j], false, false);

    async_wait_all();     // this wave's async writes into buf^1 have landed
    __syncthreads();      // everyone done reading buf / writing buf^1
  }

  // C/D layout: VGPR v, lanes 0-15 -> row v; lanes 16-31 -> row v+8
  const int rSel = (lane < 16) ? 0 : 8;
  if (bm + BM <= M) {
    // fast path: whole block tile in range, no per-row EXEC churn
#pragma unroll
    for (int i = 0; i < 2; ++i) {
#pragma unroll
      for (int j = 0; j < 2; ++j) {
        int c = bn + wn + j * 16 + laneRow;
        float bv = bias ? bias[c] : 0.0f;
        size_t base = (size_t)(bm + wm + i * 16 + rSel) * O + c;
#pragma unroll
        for (int v = 0; v < 8; ++v) {
          float val = acc[i][j][v] + bv;
          if (relu) val = val > 0.0f ? val : 0.0f;
          C[base + (size_t)v * O] = val;
        }
      }
    }
  } else {
#pragma unroll
    for (int i = 0; i < 2; ++i) {
#pragma unroll
      for (int j = 0; j < 2; ++j) {
        int c = bn + wn + j * 16 + laneRow;
        float bv = bias ? bias[c] : 0.0f;
#pragma unroll
        for (int v = 0; v < 8; ++v) {
          int r = bm + wm + i * 16 + rSel + v;
          if (r < M) {
            float val = acc[i][j][v] + bv;
            if (relu) val = val > 0.0f ? val : 0.0f;
            C[(size_t)r * O + c] = val;
          }
        }
      }
    }
  }
}

// ---------------------------------------------------------------------------
// Elementwise / pack kernels
// ---------------------------------------------------------------------------
__global__ void k_concat2_f32(const float* __restrict__ a, const float* __restrict__ b,
                              float* __restrict__ out, int D1, int D2, int n) {
  long long t = (long long)blockIdx.x * blockDim.x + threadIdx.x;
  int Dt = D1 + D2;
  if (t >= (long long)n * Dt) return;
  int row = (int)(t / Dt), col = (int)(t % Dt);
  out[t] = (col < D1) ? a[(size_t)row * D1 + col] : b[(size_t)row * D2 + (col - D1)];
}

__global__ void k_pack2_bf16(const float* __restrict__ a, const float* __restrict__ b,
                             unsigned short* __restrict__ out, int D1, int D2, int n) {
  long long t = (long long)blockIdx.x * blockDim.x + threadIdx.x;
  int Dt = D1 + D2;
  if (t >= (long long)n * Dt) return;
  int row = (int)(t / Dt), col = (int)(t % Dt);
  float v = (col < D1) ? a[(size_t)row * D1 + col] : b[(size_t)row * D2 + (col - D1)];
  out[t] = f2bf(v);
}

__global__ void k_cvt_bf16(const float* __restrict__ in, unsigned short* __restrict__ out, int n) {
  int t = blockIdx.x * blockDim.x + threadIdx.x;
  if (t < n) out[t] = f2bf(in[t]);
}

__global__ void k_zero_f32(float* p, int n) {
  int t = blockIdx.x * blockDim.x + threadIdx.x;
  if (t < n) p[t] = 0.0f;
}
__global__ void k_fill_i32(int* p, int v, int n) {
  int t = blockIdx.x * blockDim.x + threadIdx.x;
  if (t < n) p[t] = v;
}

// ---------------------------------------------------------------------------
// Attention: per-node dots a_s[v]=z[v].attn[:D], a_d[v]=z[v].attn[D:]
// One wave32 per node, shuffle reduce.
// ---------------------------------------------------------------------------
__global__ __launch_bounds__(256) void k_attn_dots(
    const float* __restrict__ z, const float* __restrict__ attn,
    float* __restrict__ a_s, float* __restrict__ a_d, int D, int n)
{
  int node = blockIdx.x * 8 + (threadIdx.x >> 5);
  int lane = threadIdx.x & 31;
  if (node >= n) return;
  const float* zr = z + (size_t)node * D;
  float s0 = 0.0f, s1 = 0.0f;
  for (int c = lane; c < D; c += 32) {
    float zz = zr[c];
    s0 += zz * attn[c];
    s1 += zz * attn[D + c];
  }
#pragma unroll
  for (int off = 16; off > 0; off >>= 1) {
    s0 += __shfl_down(s0, off, 32);
    s1 += __shfl_down(s1, off, 32);
  }
  if (lane == 0) { a_s[node] = s0; a_d[node] = s1; }
}

__global__ void k_edge_max(const int* __restrict__ src, const int* __restrict__ dst,
                           const float* __restrict__ a_s, const float* __restrict__ a_d,
                           float* __restrict__ ebuf, int* __restrict__ smax, int E_)
{
  int e = blockIdx.x * blockDim.x + threadIdx.x;
  if (e >= E_) return;
  float v = a_s[src[e]] + a_d[dst[e]];
  v = (v >= 0.0f) ? v : 0.2f * v;               // leaky relu
  ebuf[e] = v;
  atomicMax(&smax[dst[e]], f2ord(v));
}

__global__ void k_edge_exp(const int* __restrict__ dst, float* __restrict__ ebuf,
                           const int* __restrict__ smax, float* __restrict__ den, int E_)
{
  int e = blockIdx.x * blockDim.x + threadIdx.x;
  if (e >= E_) return;
  int d = dst[e];
  int im = smax[d];
  float m = (im == (int)0x80000000) ? 0.0f : ord2f(im);
  float ex = __expf(ebuf[e] - m);
  ebuf[e] = ex;
  atomAddF(&den[d], ex);
}

// wave32 per edge: agg[dst] += alpha * z[src]  (coalesced in feature dim)
__global__ __launch_bounds__(256) void k_edge_agg(
    const int* __restrict__ src, const int* __restrict__ dst,
    const float* __restrict__ ebuf, const float* __restrict__ den,
    const float* __restrict__ z, float* __restrict__ agg, int D, int E_)
{
  int e = blockIdx.x * 8 + (threadIdx.x >> 5);
  int lane = threadIdx.x & 31;
  if (e >= E_) return;
  int s = src[e], d = dst[e];
  float dn = den[d];
  float w = ebuf[e] / (dn > 0.0f ? dn : 1.0f);
  const float* zr = z + (size_t)s * D;
  float* ar = agg + (size_t)d * D;
  for (int c = lane; c < D; c += 32)
    atomAddF(&ar[c], w * zr[c]);
}

// ---------------------------------------------------------------------------
// Final per-node MLPs: sf = x@src_mlp_w+b, df = x@dst_mlp_w+b  (D=128 -> 6)
// ---------------------------------------------------------------------------
__global__ void k_node_mlp(const float* __restrict__ x,
                           const float* __restrict__ sw, const float* __restrict__ sb,
                           const float* __restrict__ dw, const float* __restrict__ db,
                           float* __restrict__ sf, float* __restrict__ df, int D, int n)
{
  long long t = (long long)blockIdx.x * blockDim.x + threadIdx.x;
  if (t >= (long long)n * 12) return;
  int node = (int)(t / 12);
  int j = (int)(t % 12);
  const float* xr = x + (size_t)node * D;
  if (j < 6) {
    float a = sb[j];
    for (int k = 0; k < D; ++k) a += xr[k] * sw[k * 6 + j];
    sf[node * 6 + j] = a;
  } else {
    j -= 6;
    float a = db[j];
    for (int k = 0; k < D; ++k) a += xr[k] * dw[k * 6 + j];
    df[node * 6 + j] = a;
  }
}

// ---------------------------------------------------------------------------
// Edge classifier + density message accumulation
// ---------------------------------------------------------------------------
__global__ __launch_bounds__(256) void k_edge_cls(
    const int* __restrict__ src, const int* __restrict__ dst,
    const float* __restrict__ sf, const float* __restrict__ df,
    const float* __restrict__ xws, const float* __restrict__ yws,
    const float* __restrict__ raw,
    const float* __restrict__ p1, const float* __restrict__ w1, const float* __restrict__ b1,
    const float* __restrict__ p2, const float* __restrict__ w2, const float* __restrict__ b2,
    float* __restrict__ pred_conn, float* __restrict__ sacc, float* __restrict__ cacc, int E_)
{
  __shared__ float sp1[16], sw1[128], sb1[8], sp2[8], sw2[8], sb2[1];
  int tid = threadIdx.x;
  if (tid < 16)  sp1[tid] = p1[tid];
  if (tid < 128) sw1[tid] = w1[tid];
  if (tid < 8) { sb1[tid] = b1[tid]; sp2[tid] = p2[tid]; sw2[tid] = w2[tid]; }
  if (tid == 0) sb2[0] = b2[0];
  __syncthreads();

  int e = blockIdx.x * blockDim.x + tid;
  if (e >= E_) return;
  int s = src[e], d = dst[e];

  float f[16];
#pragma unroll
  for (int j = 0; j < 6; ++j) f[j] = sf[s * 6 + j];
  f[6] = xws[s]; f[7] = yws[s];
#pragma unroll
  for (int j = 0; j < 6; ++j) f[8 + j] = df[d * 6 + j];
  f[14] = xws[d]; f[15] = yws[d];
#pragma unroll
  for (int j = 0; j < 16; ++j) f[j] = (f[j] >= 0.0f) ? f[j] : sp1[j] * f[j];

  float h[8];
#pragma unroll
  for (int o = 0; o < 8; ++o) {
    float a = sb1[o];
#pragma unroll
    for (int j = 0; j < 16; ++j) a += f[j] * sw1[j * 8 + o];
    h[o] = (a >= 0.0f) ? a : sp2[o] * a;
  }
  float pc = sb2[0];
#pragma unroll
  for (int o = 0; o < 8; ++o) pc += h[o] * sw2[o];

  pred_conn[e] = pc;
  float prob = 1.0f / (1.0f + __expf(-pc));
  float msg = raw[e] * (2.0f * prob - 1.0f);
  atomAddF(&sacc[s], msg);
  atomAddF(&cacc[s], 1.0f);
}

__global__ void k_den_final(const float* __restrict__ sacc, const float* __restrict__ cacc,
                            float* __restrict__ out, int n)
{
  int v = blockIdx.x * blockDim.x + threadIdx.x;
  if (v >= n) return;
  float c = cacc[v];
  out[v] = sacc[v] / (c > 0.0f ? c : 1.0f);
}

// ---------------------------------------------------------------------------
// Host orchestration (graph-capture safe: only kernel launches on `stream`)
// ---------------------------------------------------------------------------
static inline dim3 g1(long long n, int b) { return dim3((unsigned)((n + b - 1) / b)); }

extern "C" void kernel_launch(void* const* d_in, const int* in_sizes, int n_in,
                              void* d_out, int out_size, void* d_ws, size_t ws_size,
                              hipStream_t stream)
{
  const int N = NN, E = EE;

  // Input layout: setup_inputs() insertion order; `params` expands to its
  // jax tree leaves (dict keys sorted alphabetically; convs list in order,
  // each conv dict sorted: attn_w, fc_w, nn_b, nn_w).
  const float* features = (const float*)d_in[0];
  const float* clusterf = (const float*)d_in[1];
  const float* xws      = (const float*)d_in[2];
  const float* yws      = (const float*)d_in[3];
  const float* raw      = (const float*)d_in[4];
  const float* cls1_b   = (const float*)d_in[5];
  const float* cls1_w   = (const float*)d_in[6];
  const float* cls2_b   = (const float*)d_in[7];
  const float* cls2_w   = (const float*)d_in[8];
  const float* attn_w[4]; const float* fc_w[4]; const float* nn_b[4]; const float* nn_w[4];
  for (int c = 0; c < 4; ++c) {
    attn_w[c] = (const float*)d_in[9 + 4 * c + 0];
    fc_w[c]   = (const float*)d_in[9 + 4 * c + 1];
    nn_b[c]   = (const float*)d_in[9 + 4 * c + 2];
    nn_w[c]   = (const float*)d_in[9 + 4 * c + 3];
  }
  const float* dst_mlp_b = (const float*)d_in[25];
  const float* dst_mlp_w = (const float*)d_in[26];
  const float* prelu1_a  = (const float*)d_in[27];
  const float* prelu2_a  = (const float*)d_in[28];
  const float* src_mlp_b = (const float*)d_in[29];
  const float* src_mlp_w = (const float*)d_in[30];
  const int* src = (const int*)d_in[31];
  const int* dst = (const int*)d_in[32];
  float* out = (float*)d_out;   // [E] pred_conn, then [N] pred_den

  // Workspace carve (256B aligned slices)
  char* p = (char*)d_ws;
  auto carve = [&](size_t bytes) -> void* {
    void* r = (void*)p;
    p += (bytes + 255) & ~(size_t)255;
    return r;
  };
  float* xA  = (float*)carve((size_t)N * 256 * 4);
  float* xB  = (float*)carve((size_t)N * 256 * 4);
  float* z   = (float*)carve((size_t)N * 256 * 4);
  float* agg = (float*)carve((size_t)N * 256 * 4);
  unsigned short* Ab = (unsigned short*)carve((size_t)N * 512 * 2);
  unsigned short* Wb = (unsigned short*)carve((size_t)512 * 256 * 2);
  float* a_s  = (float*)carve((size_t)N * 4);
  float* a_d  = (float*)carve((size_t)N * 4);
  int*   smax = (int*)carve((size_t)N * 4);
  float* den  = (float*)carve((size_t)N * 4);
  float* ebuf = (float*)carve((size_t)E * 4);
  float* sf   = (float*)carve((size_t)N * 6 * 4);
  float* df   = (float*)carve((size_t)N * 6 * 4);
  float* sacc = (float*)carve((size_t)N * 4);
  float* cacc = (float*)carve((size_t)N * 4);
  if ((size_t)(p - (char*)d_ws) > ws_size) return;  // not enough scratch

  dim3 blk(256);

  // x0 = concat(features, cluster_features)  [N,256]
  k_concat2_f32<<<g1((long long)N * 256, 256), blk, 0, stream>>>(features, clusterf, xA, 128, 128, N);

  const int idim[4] = {256, 256, 256, 128};
  const int odim[4] = {256, 256, 128, 128};
  float* xcur = xA; float* xnext = xB;

  for (int c = 0; c < 4; ++c) {
    const int D = idim[c], O = odim[c];

    // z = x @ fc_w   [N,D]
    k_cvt_bf16<<<g1((long long)N * D, 256), blk, 0, stream>>>(xcur, Ab, N * D);
    k_cvt_bf16<<<g1((long long)D * D, 256), blk, 0, stream>>>(fc_w[c], Wb, D * D);
    dim3 gz((N + BM - 1) / BM, D / BN);
    k_gemm_bf16<<<gz, blk, 0, stream>>>(Ab, Wb, nullptr, z, N, D, D, 0);

    // attention scores and segment softmax over dst
    k_attn_dots<<<dim3((N + 7) / 8), blk, 0, stream>>>(z, attn_w[c], a_s, a_d, D, N);
    k_fill_i32<<<g1(N, 256), blk, 0, stream>>>(smax, (int)0x80000000, N);
    k_zero_f32<<<g1(N, 256), blk, 0, stream>>>(den, N);
    k_zero_f32<<<g1((long long)N * D, 256), blk, 0, stream>>>(agg, N * D);
    k_edge_max<<<g1(E, 256), blk, 0, stream>>>(src, dst, a_s, a_d, ebuf, smax, E);
    k_edge_exp<<<g1(E, 256), blk, 0, stream>>>(dst, ebuf, smax, den, E);
    k_edge_agg<<<dim3((E + 7) / 8), blk, 0, stream>>>(src, dst, ebuf, den, z, agg, D, E);

    // h = relu([x, agg] @ nn_w + nn_b)   [N,O]
    k_pack2_bf16<<<g1((long long)N * 2 * D, 256), blk, 0, stream>>>(xcur, agg, Ab, D, D, N);
    k_cvt_bf16<<<g1((long long)2 * D * O, 256), blk, 0, stream>>>(nn_w[c], Wb, 2 * D * O);
    dim3 gh((N + BM - 1) / BM, O / BN);
    k_gemm_bf16<<<gh, blk, 0, stream>>>(Ab, Wb, nn_b[c], xnext, N, 2 * D, O, 1);

    float* t = xcur; xcur = xnext; xnext = t;
  }

  // pred_conn + density
  k_node_mlp<<<g1((long long)N * 12, 256), blk, 0, stream>>>(
      xcur, src_mlp_w, src_mlp_b, dst_mlp_w, dst_mlp_b, sf, df, 128, N);
  k_zero_f32<<<g1(N, 256), blk, 0, stream>>>(sacc, N);
  k_zero_f32<<<g1(N, 256), blk, 0, stream>>>(cacc, N);
  k_edge_cls<<<g1(E, 256), blk, 0, stream>>>(
      src, dst, sf, df, xws, yws, raw,
      prelu1_a, cls1_w, cls1_b, prelu2_a, cls2_w, cls2_b,
      out, sacc, cacc, E);
  k_den_final<<<g1(N, 256), blk, 0, stream>>>(sacc, cacc, out + E, N);
}
